// NRDF_Adapter_7009386627272
// MI455X (gfx1250) — matrix operand order repaired and checked by reading, not compiled, not run on hardware
//
#include <hip/hip_runtime.h>
#include <hip/hip_bf16.h>

typedef __attribute__((ext_vector_type(16))) __bf16 v16bf;
typedef __attribute__((ext_vector_type(8)))  float  v8f;

#define J_ 29
#define F_ 16
#define H_ 17
#define BTILE 128
#define KPAD0 480                 // 464 padded to 15*32
#define SMEM_A (BTILE * KPAD0 * 2)   // 122880 B : latent, later h2
#define SMEM_B (BTILE * 512 * 2)     // 131072 B : h1, later h3

__constant__ int c_topo[J_] = {12,0,6,13,1,7,14,2,8,15,22,3,9,16,23,4,10,17,24,5,11,18,25,19,26,20,27,21,28};
__constant__ int c_parent[J_] = {12,0,1,2,3,4,12,6,7,8,9,10,-1,12,13,14,15,16,17,18,19,20,14,22,23,24,25,26,27};

__device__ __forceinline__ float softplus100(float z) {
  // logaddexp(100 z, 0)/100 = max(t,0) + log1p(exp(-|t|)), t = 100 z
  float t = z * 100.0f;
  float e = __expf(-fabsf(t));
  return (fmaxf(t, 0.0f) + __logf(1.0f + e)) * 0.01f;
}

// One GEMM layer: D(128 x N) = softplus(A(128 x K) * W^T + bias), all tiles 16x16x32 bf16 WMMA.
// A in LDS (row-major, lda elems), W in global bf16 (row-major [N][K]), D in LDS.
// 4 waves, wave handles M-tiles {2w, 2w+1}; A-fragments kept resident in VGPRs.
template<int KSTEPS, int NTILES>
__device__ __forceinline__ void gemm_bf16(const __bf16* __restrict__ A, int lda,
                                          const __bf16* __restrict__ W,
                                          const float* __restrict__ bias,
                                          __bf16* __restrict__ D, int ldd,
                                          int wave, int lane) {
  const int K  = KSTEPS * 32;
  const int nl = lane & 15;            // N within tile (C/D layout)
  const int mb = (lane >> 4) << 3;     // M base: lanes 16-31 hold M=8..15
  const int arow = lane & 15;          // A: M = lane%16
  const int akb  = (lane < 16) ? 0 : 16;  // A: hi lanes hold K=16..31 of each 32-slice
  const int bkb  = (lane < 16) ? 0 : 16;  // B: hi lanes hold K=16..31 of each 32-slice

  for (int mi = 0; mi < 2; ++mi) {
    const int mt = wave * 2 + mi;
    v16bf afr[KSTEPS];
#pragma unroll
    for (int ks = 0; ks < KSTEPS; ++ks) {
      const __bf16* p = A + (mt * 16 + arow) * lda + ks * 32 + akb;
#pragma unroll
      for (int e = 0; e < 16; ++e) afr[ks][e] = p[e];        // K = akb+e
    }
    for (int nt = 0; nt < NTILES; ++nt) {
      v8f acc = {};
#pragma unroll
      for (int ks = 0; ks < KSTEPS; ++ks) {
        const __bf16* q = W + (size_t)(nt * 16 + nl) * K + ks * 32 + bkb;
        v16bf bfr;
#pragma unroll
        for (int e = 0; e < 16; ++e) bfr[e] = q[e];
        acc = __builtin_amdgcn_wmma_f32_16x16x32_bf16(false, afr[ks], false, bfr,
                                                      (short)0, acc, false, false);
      }
      float b = bias[nt * 16 + nl];
#pragma unroll
      for (int r = 0; r < 8; ++r) {
        float v = softplus100(acc[r] + b);
        D[(mt * 16 + mb + r) * ldd + nt * 16 + nl] = (__bf16)v;
      }
    }
  }
}

// Pre-convert decoder weights fp32 -> bf16 into workspace, padding Wd0's K 464->480 with zeros.
#define N0 (512 * 480)
#define N1 (256 * 512)
#define N2 (128 * 256)
__global__ void convert_weights(const float* __restrict__ Wd0, const float* __restrict__ Wd1,
                                const float* __restrict__ Wd2, __bf16* __restrict__ ws) {
  int idx = blockIdx.x * 256 + threadIdx.x;
  if (idx < N0) {
    int n = idx / KPAD0, k = idx - n * KPAD0;
    ws[idx] = (k < 464) ? (__bf16)Wd0[n * 464 + k] : (__bf16)0.0f;
  } else if (idx < N0 + N1) {
    int t = idx - N0;
    ws[idx] = (__bf16)Wd1[t];
  } else if (idx < N0 + N1 + N2) {
    int t = idx - N0 - N1;
    ws[idx] = (__bf16)Wd2[t];
  }
}

__global__ __launch_bounds__(BTILE) void nrdf_fused(
    const float* __restrict__ x,
    const float* __restrict__ W1, const float* __restrict__ b1,
    const float* __restrict__ W2, const float* __restrict__ b2,
    const __bf16* __restrict__ Wd0bf, const float* __restrict__ bd0,
    const __bf16* __restrict__ Wd1bf, const float* __restrict__ bd1,
    const __bf16* __restrict__ Wd2bf, const float* __restrict__ bd2,
    const float* __restrict__ Wd3, const float* __restrict__ bd3,
    float* __restrict__ out, int Btotal) {
  __shared__ __align__(16) unsigned char smem[SMEM_A + SMEM_B];
  __bf16* lat = (__bf16*)smem;              // [128][480] bf16
  __bf16* h1  = (__bf16*)(smem + SMEM_A);   // [128][512]
  __bf16* h2  = (__bf16*)smem;              // [128][256] (reuses latent region)
  __bf16* h3  = (__bf16*)(smem + SMEM_A);   // [128][128] (reuses h1 region)

  const int tid  = threadIdx.x;
  const int lane = tid & 31;
  const int wave = tid >> 5;
  const int row  = blockIdx.x * BTILE + tid;
  const bool valid = row < Btotal;

  // ---------- Stage 1: kinematic-tree joint MLPs, one sample row per thread ----------
  {
    __bf16* lrow = lat + tid * KPAD0;
#pragma unroll
    for (int c = 0; c < 16; ++c) lrow[464 + c] = (__bf16)0.0f;  // K padding
    if (valid) {
      for (int jj = 0; jj < J_; ++jj) {
        const int i = c_topo[jj];       // wave-uniform -> scalar loads for weights
        const int p = c_parent[i];
        float inp[H_];
        inp[0] = x[(size_t)row * J_ + i];
        if (p >= 0) {
#pragma unroll
          for (int f = 0; f < F_; ++f) inp[1 + f] = (float)lrow[p * F_ + f];
        } else {
#pragma unroll
          for (int f = 0; f < F_; ++f) inp[1 + f] = 0.0f;
        }
        float h[H_];
#pragma unroll
        for (int j = 0; j < H_; ++j) {
          float a = b1[i * H_ + j];
          const float* w = W1 + ((size_t)i * H_ + j) * H_;
#pragma unroll
          for (int k = 0; k < H_; ++k) a = fmaf(w[k], inp[k], a);
          h[j] = softplus100(a);
        }
#pragma unroll
        for (int f = 0; f < F_; ++f) {
          float a = b2[i * F_ + f];
          const float* w = W2 + ((size_t)i * F_ + f) * H_;
#pragma unroll
          for (int j = 0; j < H_; ++j) a = fmaf(w[j], h[j], a);
          lrow[i * F_ + f] = (__bf16)softplus100(a);
        }
      }
    } else {
      for (int c = 0; c < 464; ++c) lrow[c] = (__bf16)0.0f;
    }
  }
  __syncthreads();

  // ---------- Decoder via bf16 WMMA (f32 accumulate) ----------
  gemm_bf16<15, 32>(lat, KPAD0, Wd0bf, bd0, h1, 512, wave, lane);  // 464(480)->512
  __syncthreads();
  gemm_bf16<16, 16>(h1, 512, Wd1bf, bd1, h2, 256, wave, lane);     // 512->256
  __syncthreads();
  gemm_bf16<8, 8>(h2, 256, Wd2bf, bd2, h3, 128, wave, lane);       // 256->128
  __syncthreads();

  // ---------- Final 128 -> 1 : per-thread dot product ----------
  if (valid) {
    float a = bd3[0];
    const __bf16* hr = h3 + tid * 128;
#pragma unroll 8
    for (int k = 0; k < 128; ++k) a = fmaf((float)hr[k], Wd3[k], a);
    out[row] = softplus100(a);
  }
}

extern "C" void kernel_launch(void* const* d_in, const int* in_sizes, int n_in,
                              void* d_out, int out_size, void* d_ws, size_t ws_size,
                              hipStream_t stream) {
  const float* x   = (const float*)d_in[0];
  const float* W1  = (const float*)d_in[1];
  const float* b1  = (const float*)d_in[2];
  const float* W2  = (const float*)d_in[3];
  const float* b2  = (const float*)d_in[4];
  const float* Wd0 = (const float*)d_in[5];
  const float* bd0 = (const float*)d_in[6];
  const float* Wd1 = (const float*)d_in[7];
  const float* bd1 = (const float*)d_in[8];
  const float* Wd2 = (const float*)d_in[9];
  const float* bd2 = (const float*)d_in[10];
  const float* Wd3 = (const float*)d_in[11];
  const float* bd3 = (const float*)d_in[12];

  const int B = in_sizes[0] / J_;
  __bf16* wsbf = (__bf16*)d_ws;
  const int total = N0 + N1 + N2;  // 409600 bf16 = 819200 B in d_ws

  convert_weights<<<(total + 255) / 256, 256, 0, stream>>>(Wd0, Wd1, Wd2, wsbf);

  const int nblk = (B + BTILE - 1) / BTILE;  // B=131072 -> 1024 workgroups
  nrdf_fused<<<nblk, BTILE, 0, stream>>>(x, W1, b1, W2, b2,
                                         wsbf, bd0,
                                         wsbf + N0, bd1,
                                         wsbf + N0 + N1, bd2,
                                         Wd3, bd3,
                                         (float*)d_out, B);
}
